// CoverageAttnDecoderRNN_20547123544564
// MI455X (gfx1250) — matrix operand order repaired and checked by reading, not compiled, not run on hardware
//
#include <hip/hip_runtime.h>
#include <hip/hip_bf16.h>
#include <stddef.h>

// ---------------- problem constants ----------------
#define BB   128
#define LL   400
#define HH   1024
#define EE   512
#define VV   50000
#define EXT  51000
#define TPREV 10
#define H3  (3*HH)

typedef __attribute__((ext_vector_type(16))) __bf16          v16bf;
typedef __attribute__((ext_vector_type(16))) unsigned short  v16u;
typedef __attribute__((ext_vector_type(8)))  unsigned short  v8u;
typedef __attribute__((ext_vector_type(4)))  unsigned short  v4u;
typedef __attribute__((ext_vector_type(8)))  float           v8f;
typedef __attribute__((ext_vector_type(4)))  float           v4f;

static __device__ __forceinline__ unsigned short f2bf(float f) {
    union { float f; unsigned int u; } x; x.f = f;
    unsigned int u = x.u;
    u += 0x7FFFu + ((u >> 16) & 1u);   // round to nearest even
    return (unsigned short)(u >> 16);
}
static __device__ __forceinline__ float sigm(float x) { return 1.f / (1.f + __expf(-x)); }

static __device__ __forceinline__ v16bf cat16(v8u a, v8u b) {
    v16u u = __builtin_shufflevector(a, b, 0,1,2,3,4,5,6,7,8,9,10,11,12,13,14,15);
    return __builtin_bit_cast(v16bf, u);
}

// ---------------- f32 -> bf16 conversion (vectorized, one pass per operand) ----------------
__global__ void cvt_bf16(const float* __restrict__ s, unsigned short* __restrict__ d, int n4)
{
    for (int i = blockIdx.x * blockDim.x + threadIdx.x; i < n4; i += gridDim.x * blockDim.x) {
        v4f f = ((const v4f*)s)[i];
        v4u u;
        #pragma unroll
        for (int j = 0; j < 4; ++j) u[j] = f2bf(f[j]);
        ((v4u*)d)[i] = u;
    }
}

// drop-tail strided convert: src rows have sstride cols (f32), dst rows have cols (bf16)
__global__ void cvt_bf16_strided(const float* __restrict__ s, unsigned short* __restrict__ d,
                                 int rows, int cols, int sstride)
{
    int c4n = cols >> 2;
    int total = rows * c4n;
    for (int i = blockIdx.x * blockDim.x + threadIdx.x; i < total; i += gridDim.x * blockDim.x) {
        int r = i / c4n, c4 = i - r * c4n;
        const float* sp = s + (size_t)r * sstride + c4 * 4;
        v4u u;
        #pragma unroll
        for (int j = 0; j < 4; ++j) u[j] = f2bf(sp[j]);
        *(v4u*)(d + (size_t)r * cols + c4 * 4) = u;
    }
}

// ---------------- GEMM: Out[M,N] = X[M,K]*W[N,K]^T + bias,  bf16 WMMA, f32 accum ----------
// Each wave owns one 16-row strip and FOUR 16-col tiles (shared A fragment).
// Optional f32 tail for K..Kfull-1 (odd K for outh_w).
__global__ __launch_bounds__(256)
void gemm_bf16(const unsigned short* __restrict__ X, const unsigned short* __restrict__ W,
               const float* __restrict__ bias, float* __restrict__ Out,
               int M, int N, int K,
               const float* __restrict__ Xf, const float* __restrict__ Wf, int Kfull)
{
    int wave   = (int)((blockIdx.x * blockDim.x + threadIdx.x) >> 5);
    int lane   = (int)(threadIdx.x & 31);
    int tilesN = N >> 4;
    int groupsN = (tilesN + 3) >> 2;
    int nGroups = (M >> 4) * groupsN;
    if (wave >= nGroups) return;                 // wave-uniform: EXEC stays all-ones
    int m0 = (wave / groupsN) << 4;
    int g  = wave % groupsN;
    int hi = lane >> 4, lo = lane & 15;

    int nt[4];
    #pragma unroll
    for (int t = 0; t < 4; ++t) {
        int tn = g * 4 + t;
        if (tn > tilesN - 1) tn = tilesN - 1;    // clamp: duplicate tile writes same values
        nt[t] = tn << 4;
    }

    const unsigned short* xp = X + (size_t)(m0 + lo) * K;     // A row: M = lane&15
    const unsigned short* wp[4];
    #pragma unroll
    for (int t = 0; t < 4; ++t) wp[t] = W + (size_t)(nt[t] + lo) * K;  // B col = W row

    v8f c[4] = {{}, {}, {}, {}};
    for (int k0 = 0; k0 < K; k0 += 32) {
        // A fragment (16x32 bf16): e<8 -> K=k0+hi*8+e ; e>=8 -> K=k0+16+hi*8+(e-8)
        v8u a0 = *(const v8u*)(xp + k0 + hi * 8);
        v8u a1 = *(const v8u*)(xp + k0 + 16 + hi * 8);
        v16bf a = cat16(a0, a1);
        #pragma unroll
        for (int t = 0; t < 4; ++t) {
            __builtin_prefetch((const void*)(wp[t] + k0 + 64), 0, 0);
            // B fragment (32x16 bf16): e<8 -> K=k0+hi*8+e ; e>=8 -> K=k0+16+hi*8+(e-8)
            v8u b0 = *(const v8u*)(wp[t] + k0 + hi * 8);
            v8u b1 = *(const v8u*)(wp[t] + k0 + 16 + hi * 8);
            v16bf b = cat16(b0, b1);
            c[t] = __builtin_amdgcn_wmma_f32_16x16x32_bf16(false, a, false, b,
                                                           (short)0, c[t], false, false);
        }
    }

    #pragma unroll
    for (int t = 0; t < 4; ++t) {
        int n = nt[t] + lo;
        float bv = bias ? bias[n] : 0.f;
        #pragma unroll
        for (int r = 0; r < 8; ++r) {
            int m = m0 + r + 8 * hi;             // C/D: row = r + 8*hi, col = lane&15
            float acc = c[t][r];
            if (Xf)                              // f32 tail for odd K
                for (int k = K; k < Kfull; ++k)
                    acc += Xf[(size_t)m * Kfull + k] * Wf[(size_t)n * Kfull + k];
            Out[(size_t)m * N + n] = acc + bv;
        }
    }
}

// ---------------- small fused kernels ----------------
__global__ void gather_emb(const int* __restrict__ tok, const float* __restrict__ emb,
                           float* __restrict__ x, unsigned short* __restrict__ xb)
{
    int b = blockIdx.x;
    float v = emb[(size_t)tok[b] * EE + threadIdx.x];
    x [(size_t)b * EE + threadIdx.x] = v;
    xb[(size_t)b * EE + threadIdx.x] = f2bf(v);
}

__global__ void gru_combine(const float* __restrict__ gi, const float* __restrict__ gh,
                            const float* __restrict__ hprev, float* __restrict__ dec_h)
{
    int i = blockIdx.x * blockDim.x + threadIdx.x;
    if (i >= BB * HH) return;
    int b = i >> 10, h = i & (HH - 1);
    const float* gib = gi + (size_t)b * H3;
    const float* ghb = gh + (size_t)b * H3;
    float r = sigm(gib[h] + ghb[h]);
    float z = sigm(gib[HH + h] + ghb[HH + h]);
    float n = tanhf(gib[2 * HH + h] + r * ghb[2 * HH + h]);
    dec_h[i] = (1.f - z) * n + z * hprev[i];
}

__global__ void copy_f32(const float* __restrict__ s, float* __restrict__ d, int n)
{
    for (int i = blockIdx.x * blockDim.x + threadIdx.x; i < n; i += gridDim.x * blockDim.x)
        d[i] = s[i];
}

__global__ void cov_kernel(const float* __restrict__ prev_att, const float* __restrict__ w_c,
                           float* __restrict__ cov)
{
    int i = blockIdx.x * blockDim.x + threadIdx.x;
    if (i >= BB * LL) return;
    int b = i / LL, l = i - b * LL;
    float s = 0.f;
    #pragma unroll
    for (int t = 0; t < TPREV; ++t) s += prev_att[((size_t)b * TPREV + t) * LL + l];
    cov[i] = w_c[l] * s;
}

__global__ __launch_bounds__(256)
void score_kernel(const float* __restrict__ enc, const float* __restrict__ dec_h,
                  const float* __restrict__ w_h, const float* __restrict__ w_s,
                  const float* __restrict__ attn_v, const float* __restrict__ cov,
                  const float* __restrict__ att_bias, float* __restrict__ scores)
{
    int bl = blockIdx.x;
    int b = bl / LL;
    const float* e  = enc + (size_t)bl * HH;
    const float* dh = dec_h + (size_t)b * HH;
    float cv = cov[bl] + att_bias[0];
    float acc = 0.f;
    for (int h = threadIdx.x; h < HH; h += 256)
        acc += attn_v[h] * tanhf(w_h[h] * e[h] + w_s[h] * dh[h] + cv);
    __shared__ float red[256];
    red[threadIdx.x] = acc; __syncthreads();
    for (int s = 128; s > 0; s >>= 1) {
        if (threadIdx.x < (unsigned)s) red[threadIdx.x] += red[threadIdx.x + s];
        __syncthreads();
    }
    if (threadIdx.x == 0) scores[bl] = red[0];
}

__global__ __launch_bounds__(256)
void softmax_L(const float* __restrict__ scores, float* __restrict__ att)
{
    int b = blockIdx.x;
    const float* s = scores + (size_t)b * LL;
    __shared__ float red[256];
    float m = -1e30f;
    for (int l = threadIdx.x; l < LL; l += 256) m = fmaxf(m, s[l]);
    red[threadIdx.x] = m; __syncthreads();
    for (int st = 128; st > 0; st >>= 1) {
        if (threadIdx.x < (unsigned)st) red[threadIdx.x] = fmaxf(red[threadIdx.x], red[threadIdx.x + st]);
        __syncthreads();
    }
    m = red[0]; __syncthreads();
    float sum = 0.f;
    for (int l = threadIdx.x; l < LL; l += 256) sum += __expf(s[l] - m);
    red[threadIdx.x] = sum; __syncthreads();
    for (int st = 128; st > 0; st >>= 1) {
        if (threadIdx.x < (unsigned)st) red[threadIdx.x] += red[threadIdx.x + st];
        __syncthreads();
    }
    float inv = 1.f / red[0];
    for (int l = threadIdx.x; l < LL; l += 256)
        att[(size_t)b * LL + l] = __expf(s[l] - m) * inv;
}

__global__ void newprev_kernel(const float* __restrict__ prev_att, const float* __restrict__ att,
                               float* __restrict__ npa)
{
    int i = blockIdx.x * blockDim.x + threadIdx.x;
    if (i >= BB * (TPREV + 1) * LL) return;
    int b = i / ((TPREV + 1) * LL);
    int r = i - b * (TPREV + 1) * LL;
    int t = r / LL, l = r - t * LL;
    npa[i] = (t < TPREV) ? prev_att[((size_t)b * TPREV + t) * LL + l]
                         : att[(size_t)b * LL + l];
}

__global__ __launch_bounds__(256)
void ctx_kernel(const float* __restrict__ att, const float* __restrict__ enc,
                float* __restrict__ ctx)
{
    int b = blockIdx.y;
    int h = blockIdx.x * 256 + threadIdx.x;
    const float* eb = enc + (size_t)b * LL * HH + h;
    const float* ab = att + (size_t)b * LL;
    float acc = 0.f;
    for (int l = 0; l < LL; ++l) acc += ab[l] * eb[(size_t)l * HH];
    ctx[(size_t)b * HH + h] = acc;
}

// writes f32 combined (stride 2049) AND bf16 combined (stride 2048, first 2048 cols)
__global__ void combined_kernel(const float* __restrict__ dec_h, const float* __restrict__ ctx,
                                const float* __restrict__ ctrl, float* __restrict__ comb,
                                unsigned short* __restrict__ combb)
{
    int i = blockIdx.x * blockDim.x + threadIdx.x;
    if (i >= BB * (2 * HH + 1)) return;
    int b = i / (2 * HH + 1), j = i - b * (2 * HH + 1);
    float v;
    if (j < HH)           v = dec_h[(size_t)b * HH + j];
    else if (j < 2 * HH)  v = ctx[(size_t)b * HH + (j - HH)];
    else                  v = ctrl[b];
    comb[i] = v;
    if (j < 2 * HH) combb[(size_t)b * 2 * HH + j] = f2bf(v);
}

__global__ __launch_bounds__(256)
void pgen_kernel(const float* __restrict__ comb, const float* __restrict__ x,
                 const float* __restrict__ gen_w, const float* __restrict__ gen_b,
                 float* __restrict__ p_gen)
{
    int b = blockIdx.x;
    float acc = 0.f;
    for (int j = threadIdx.x; j < 2 * HH + 1; j += 256)
        acc += comb[(size_t)b * (2 * HH + 1) + j] * gen_w[j];
    for (int j = threadIdx.x; j < EE; j += 256)
        acc += x[(size_t)b * EE + j] * gen_w[2 * HH + 1 + j];
    __shared__ float red[256];
    red[threadIdx.x] = acc; __syncthreads();
    for (int s = 128; s > 0; s >>= 1) {
        if (threadIdx.x < (unsigned)s) red[threadIdx.x] += red[threadIdx.x + s];
        __syncthreads();
    }
    if (threadIdx.x == 0) p_gen[b] = sigm(red[0] + gen_b[0]);
}

__global__ __launch_bounds__(512)
void softmax_V(const float* __restrict__ logits, const float* __restrict__ p_gen,
               float* __restrict__ p_vocab, float* __restrict__ p_final)
{
    int b = blockIdx.x;
    const float* lg = logits + (size_t)b * VV;
    __shared__ float red[512];
    float m = -1e30f;
    for (int v = threadIdx.x; v < VV; v += 512) m = fmaxf(m, lg[v]);
    red[threadIdx.x] = m; __syncthreads();
    for (int s = 256; s > 0; s >>= 1) {
        if (threadIdx.x < (unsigned)s) red[threadIdx.x] = fmaxf(red[threadIdx.x], red[threadIdx.x + s]);
        __syncthreads();
    }
    m = red[0]; __syncthreads();
    float sum = 0.f;
    for (int v = threadIdx.x; v < VV; v += 512) sum += __expf(lg[v] - m);
    red[threadIdx.x] = sum; __syncthreads();
    for (int s = 256; s > 0; s >>= 1) {
        if (threadIdx.x < (unsigned)s) red[threadIdx.x] += red[threadIdx.x + s];
        __syncthreads();
    }
    float inv = 1.f / red[0];
    float pg = p_gen[b];
    for (int v = threadIdx.x; v < VV; v += 512) {
        float p = __expf(lg[v] - m) * inv;
        p_vocab[(size_t)b * VV + v]  = p;
        p_final[(size_t)b * EXT + v] = p * pg;
    }
    for (int v = VV + threadIdx.x; v < EXT; v += 512)
        p_final[(size_t)b * EXT + v] = 0.f;
}

__global__ void scatter_kernel(const float* __restrict__ att, const int* __restrict__ fiv,
                               const float* __restrict__ p_gen, float* __restrict__ p_final)
{
    int i = blockIdx.x * blockDim.x + threadIdx.x;
    if (i >= BB * LL) return;
    int b = i / LL;
    float w = (1.f - p_gen[b]) * att[i];
    atomicAdd(p_final + (size_t)b * EXT + fiv[i], w);
}

// ---------------- launcher ----------------
extern "C" void kernel_launch(void* const* d_in, const int* in_sizes, int n_in,
                              void* d_out, int out_size, void* d_ws, size_t ws_size,
                              hipStream_t stream)
{
    const int*   input_token = (const int*)  d_in[0];
    const float* prev_dh     = (const float*)d_in[1];
    const float* last_hidden = (const float*)d_in[2];
    const float* enc         = (const float*)d_in[3];
    const int*   full_input  = (const int*)  d_in[4];
    const float* prev_att    = (const float*)d_in[5];
    const float* ctrl        = (const float*)d_in[6];
    const float* emb         = (const float*)d_in[7];
    const float* w_ih        = (const float*)d_in[8];
    const float* w_hh        = (const float*)d_in[9];
    const float* b_ih        = (const float*)d_in[10];
    const float* b_hh        = (const float*)d_in[11];
    const float* w_h         = (const float*)d_in[12];
    const float* w_s         = (const float*)d_in[13];
    const float* w_c         = (const float*)d_in[14];
    const float* att_bias    = (const float*)d_in[15];
    const float* attn_v      = (const float*)d_in[16];
    const float* gen_w       = (const float*)d_in[17];
    const float* gen_b       = (const float*)d_in[18];
    const float* outh_w      = (const float*)d_in[19];
    const float* outh_b      = (const float*)d_in[20];
    const float* outv_w      = (const float*)d_in[21];
    const float* outv_b      = (const float*)d_in[22];

    // d_out layout (flat, return order)
    float* out       = (float*)d_out;
    float* o_pfinal  = out;                                   // 128*51000
    float* o_pgen    = out + (size_t)BB * EXT;                // 128
    float* o_pvocab  = o_pgen + BB;                           // 128*50000
    float* o_att     = o_pvocab + (size_t)BB * VV;            // 128*400
    float* o_prevdh  = o_att + (size_t)BB * LL;               // 128*10*1024
    float* o_dech    = o_prevdh + (size_t)BB * TPREV * HH;    // 128*1024
    float* o_npa     = o_dech + (size_t)BB * HH;              // 128*11*400

    // ---------------- workspace layout ----------------
    // f32 region
    float* ws      = (float*)d_ws;
    float* w_x     = ws;                                  // 128*512
    float* w_gi    = w_x    + (size_t)BB * EE;            // 128*3072
    float* w_gh    = w_gi   + (size_t)BB * H3;            // 128*3072
    float* w_cov   = w_gh   + (size_t)BB * H3;            // 128*400
    float* w_sc    = w_cov  + (size_t)BB * LL;            // 128*400
    float* w_ctx   = w_sc   + (size_t)BB * LL;            // 128*1024
    float* w_comb  = w_ctx  + (size_t)BB * HH;            // 128*2049
    float* w_outh  = w_comb + (size_t)BB * (2 * HH + 1);  // 128*512
    float* w_logit = w_outh + (size_t)BB * EE;            // 128*50000
    // bf16 region (starts 16B-aligned after f32 region)
    unsigned short* hb      = (unsigned short*)(w_logit + (size_t)BB * VV);
    unsigned short* h_x     = hb;                                 // 128*512
    unsigned short* h_lh    = h_x    + (size_t)BB * EE;           // 128*1024
    unsigned short* h_wih   = h_lh   + (size_t)BB * HH;           // 3072*512
    unsigned short* h_whh   = h_wih  + (size_t)H3 * EE;           // 3072*1024
    unsigned short* h_comb  = h_whh  + (size_t)H3 * HH;           // 128*2048
    unsigned short* h_outhw = h_comb + (size_t)BB * 2 * HH;       // 512*2048
    unsigned short* h_outh  = h_outhw+ (size_t)EE * 2 * HH;       // 128*512
    unsigned short* h_outvw = h_outh + (size_t)BB * EE;           // 50000*512

    // ---- one-pass bf16 conversion of all static GEMM operands ----
    cvt_bf16<<<512,  256, 0, stream>>>(w_ih,   h_wih,  (H3 * EE) / 4);
    cvt_bf16<<<1024, 256, 0, stream>>>(w_hh,   h_whh,  (H3 * HH) / 4);
    cvt_bf16<<<4096, 256, 0, stream>>>(outv_w, h_outvw,(VV * EE) / 4);
    cvt_bf16_strided<<<1024, 256, 0, stream>>>(outh_w, h_outhw, EE, 2 * HH, 2 * HH + 1);
    cvt_bf16<<<128,  256, 0, stream>>>(last_hidden, h_lh, (BB * HH) / 4);

    // 1) embedding gather (f32 + bf16)
    gather_emb<<<BB, EE, 0, stream>>>(input_token, emb, w_x, h_x);

    // 2-3) GRU GEMMs (pure load_b128 + v_wmma inner loop)
    {
        int groupsN = ((H3 / 16) + 3) / 4;                     // 48
        int waves   = (BB / 16) * groupsN;                     // 384
        gemm_bf16<<<(waves * 32 + 255) / 256, 256, 0, stream>>>(
            h_x, h_wih, b_ih, w_gi, BB, H3, EE, nullptr, nullptr, EE);
        gemm_bf16<<<(waves * 32 + 255) / 256, 256, 0, stream>>>(
            h_lh, h_whh, b_hh, w_gh, BB, H3, HH, nullptr, nullptr, HH);
    }

    // 4) GRU gate fusion -> dec_h
    gru_combine<<<(BB * HH + 255) / 256, 256, 0, stream>>>(w_gi, w_gh, last_hidden, o_dech);

    // 5) passthrough copy
    copy_f32<<<1024, 256, 0, stream>>>(prev_dh, o_prevdh, BB * TPREV * HH);

    // 6-8) coverage, energy+scores, softmax over L
    cov_kernel<<<(BB * LL + 255) / 256, 256, 0, stream>>>(prev_att, w_c, w_cov);
    score_kernel<<<BB * LL, 256, 0, stream>>>(enc, o_dech, w_h, w_s, attn_v, w_cov, att_bias, w_sc);
    softmax_L<<<BB, 256, 0, stream>>>(w_sc, o_att);

    // 9) new_prev_att
    newprev_kernel<<<(BB * (TPREV + 1) * LL + 255) / 256, 256, 0, stream>>>(prev_att, o_att, o_npa);

    // 10) context
    {
        dim3 g(HH / 256, BB);
        ctx_kernel<<<g, 256, 0, stream>>>(o_att, enc, w_ctx);
    }

    // 11) combined concat (f32 stride 2049 + bf16 stride 2048)
    combined_kernel<<<(BB * (2 * HH + 1) + 255) / 256, 256, 0, stream>>>(o_dech, w_ctx, ctrl,
                                                                         w_comb, h_comb);

    // 12) outh GEMM: K=2048 WMMA + f32 tail for K index 2048
    {
        int groupsN = ((EE / 16) + 3) / 4;                     // 8
        int waves   = (BB / 16) * groupsN;                     // 64
        gemm_bf16<<<(waves * 32 + 255) / 256, 256, 0, stream>>>(
            h_comb, h_outhw, outh_b, w_outh, BB, EE, 2 * HH, w_comb, outh_w, 2 * HH + 1);
    }
    cvt_bf16<<<64, 256, 0, stream>>>(w_outh, h_outh, (BB * EE) / 4);

    // 13) vocab logits GEMM: 128x50000, K=512 — dominant WMMA workload
    {
        int groupsN = ((VV / 16) + 3) / 4;                     // 782
        int waves   = (BB / 16) * groupsN;                     // 6256
        gemm_bf16<<<(waves * 32 + 255) / 256, 256, 0, stream>>>(
            h_outh, h_outvw, outv_b, w_logit, BB, VV, EE, nullptr, nullptr, EE);
    }

    // 14) p_gen
    pgen_kernel<<<BB, 256, 0, stream>>>(w_comb, w_x, gen_w, gen_b, o_pgen);

    // 15) vocab softmax + p_final init
    softmax_V<<<BB, 512, 0, stream>>>(w_logit, o_pgen, o_pvocab, o_pfinal);

    // 16) pointer-generator scatter-add
    scatter_kernel<<<(BB * LL + 255) / 256, 256, 0, stream>>>(o_att, full_input, o_pgen, o_pfinal);
}
